// SpikingLayer_45466523796142
// MI455X (gfx1250) — matrix operand order repaired and checked by reading, not compiled, not run on hardware
//
#include <hip/hip_runtime.h>
#include <hip/hip_bf16.h>
#include <math.h>

// ---------------------------------------------------------------------------
// Spiking-LSTM layer for MI455X (gfx1250, wave32, WMMA 16x16x32 bf16).
//   B=8, S=2048, D=1024, H=1024.
//   Phase 1: bf16 conversion of x / w_ih / w_hh / w_fc (f32 masters kept).
//   Phase 2: x_ih = x @ w_ih^T + b_ih  (16384x4096x1024 WMMA GEMM, f32 out).
//            Register-blocked: each wave computes a 32x64 tile (8 WMMAs per
//            K-chunk, 1.5 b128 loads per WMMA).
//   Phase 3: 2048 sequential steps; per step:
//       A) gates = x_ih[t] + mem @ w_hh^T + b_hh  -> LSTM cell -> syn, mem, spk
//          (mem bf16 staged into LDS once per WG via global_load_async_to_lds,
//           ASYNCcnt-tracked, then A-fragments come from LDS)
//       B) out[t] = Dyt( spk @ w_fc^T + b_fc + x[t] )
//   Phase 4: append final syn, mem to d_out tail.
// All matmuls use v_wmma_f32_16x16x32_bf16 with f32 accumulation; recurrent
// f32 state is the master copy, bf16 is only the matmul operand (spk is {0,1},
// exact in bf16). w_hh/w_fc (10 MB bf16) stay L2-resident across all steps.
// ---------------------------------------------------------------------------

typedef __bf16 bf16_t;
typedef __attribute__((ext_vector_type(16))) __bf16 v16bf;
typedef __attribute__((ext_vector_type(8)))  __bf16 v8bf;
typedef __attribute__((ext_vector_type(8)))  float  v8f;

constexpr int Bb = 8, S = 2048, D = 1024, H = 1024;
constexpr int G4 = 4 * H;        // 4096 gate columns
constexpr int BS = Bb * S;       // 16384 rows of x
constexpr int MP = 16;           // M padded to one WMMA tile

// ---------------- WMMA helpers (layouts per CDNA5 ISA §7.12.2) ---

__device__ __forceinline__ v8f wmma_bf16(v16bf a, v16bf b, v8f c) {
  // 8 args: (neg_a, A, neg_b, B, c_mod, C, reuse_a, reuse_b)
  return __builtin_amdgcn_wmma_f32_16x16x32_bf16(false, a, false, b,
                                                 (short)0, c, false, false);
}

// A fragment: 16x32 (MxK) bf16, A row-major with leading dim lda.
// lane = half*16 + m ; element e -> K = (e&7) + (e>=8?16:0) + 8*half
__device__ __forceinline__ v16bf load_a_frag(const bf16_t* __restrict__ A,
                                             int lda, int m_base, int k_base,
                                             int lane) {
  int half = lane >> 4, m = lane & 15;
  const bf16_t* p = A + (size_t)(m_base + m) * lda + k_base + half * 8;
  v8bf lo = *reinterpret_cast<const v8bf*>(p);       // K: +0..7   (16B)
  v8bf hi = *reinterpret_cast<const v8bf*>(p + 16);  // K: +16..23 (16B)
  v16bf f;
#pragma unroll
  for (int i = 0; i < 8; ++i) { f[i] = lo[i]; f[i + 8] = hi[i]; }
  return f;
}

// B fragment: 32x16 (KxN) with Bmat[k,n] = W[n,k], W row-major N x K (ldw=K).
// lane = half*16 + n ; element e -> K = k_base + (e&7) + (e>=8?16:0) + 8*half
__device__ __forceinline__ v16bf load_b_frag(const bf16_t* __restrict__ W,
                                             int ldw, int n_base, int k_base,
                                             int lane) {
  int half = lane >> 4, n = lane & 15;
  const bf16_t* p = W + (size_t)(n_base + n) * ldw + k_base + half * 8;
  v8bf lo = *reinterpret_cast<const v8bf*>(p);
  v8bf hi = *reinterpret_cast<const v8bf*>(p + 16);
  v16bf f;
#pragma unroll
  for (int i = 0; i < 8; ++i) { f[i] = lo[i]; f[i + 8] = hi[i]; }
  return f;
}

__device__ __forceinline__ float sigmoidf_(float x) {
  return 1.0f / (1.0f + __expf(-x));
}

// Async copy of one 16-byte chunk from global to LDS (gfx1250 VGLOBAL op 98,
// tracked by ASYNCcnt).  lds_off = byte offset inside the WG's LDS allocation.
__device__ __forceinline__ void async_copy_b128(unsigned lds_off,
                                                const void* gaddr) {
  asm volatile("global_load_async_to_lds_b128 %0, %1, off"
               :: "v"(lds_off), "v"(gaddr) : "memory");
}
__device__ __forceinline__ void wait_asynccnt0() {
  asm volatile("s_wait_asynccnt 0x0" ::: "memory");
}

// ---------------- Phase 1: f32 -> bf16 conversion / zero init ---------------

__global__ void k_f32_to_bf16(const float* __restrict__ src,
                              bf16_t* __restrict__ dst, int n) {
  int i = blockIdx.x * blockDim.x + threadIdx.x;
  if (i < n) dst[i] = (bf16_t)src[i];
}

__global__ void k_zero_u32(unsigned int* __restrict__ p, int n_words) {
  int i = blockIdx.x * blockDim.x + threadIdx.x;
  if (i < n_words) p[i] = 0u;
}

// ---------------- Phase 2: x_ih = x @ w_ih^T + b_ih ------------------------
// grid = (4096/256, 16384/64), block = 256 (8 waves as 2(M) x 4(N)).
// Each wave computes a 32x64 tile: 2 A-frags x 4 B-frags -> 8 WMMAs / K-chunk.

__global__ void k_gemm_xih(const bf16_t* __restrict__ xb,
                           const bf16_t* __restrict__ wih,
                           const float* __restrict__ b_ih,
                           float* __restrict__ x_ih) {
  int lane = threadIdx.x & 31, w = threadIdx.x >> 5;
  int wr = w >> 2, wc = w & 3;
  int m0 = blockIdx.y * 64 + wr * 32;   // two 16-row tiles: m0, m0+16
  int n0 = blockIdx.x * 256 + wc * 64;  // four 16-col tiles: n0+16*i

  v8f acc[2][4] = {};
  for (int k = 0; k < D; k += 32) {
    if (k + 32 < D)  // pull next weight chunk into cache (global_prefetch_b8)
      __builtin_prefetch(wih + (size_t)(n0 + (lane & 15)) * D + k + 32, 0, 1);
    v16bf a0 = load_a_frag(xb, D, m0, k, lane);
    v16bf a1 = load_a_frag(xb, D, m0 + 16, k, lane);
    v16bf b0 = load_b_frag(wih, D, n0, k, lane);
    v16bf b1 = load_b_frag(wih, D, n0 + 16, k, lane);
    v16bf b2 = load_b_frag(wih, D, n0 + 32, k, lane);
    v16bf b3 = load_b_frag(wih, D, n0 + 48, k, lane);
    acc[0][0] = wmma_bf16(a0, b0, acc[0][0]);
    acc[0][1] = wmma_bf16(a0, b1, acc[0][1]);
    acc[0][2] = wmma_bf16(a0, b2, acc[0][2]);
    acc[0][3] = wmma_bf16(a0, b3, acc[0][3]);
    acc[1][0] = wmma_bf16(a1, b0, acc[1][0]);
    acc[1][1] = wmma_bf16(a1, b1, acc[1][1]);
    acc[1][2] = wmma_bf16(a1, b2, acc[1][2]);
    acc[1][3] = wmma_bf16(a1, b3, acc[1][3]);
  }

  int half = lane >> 4, nl = lane & 15;
#pragma unroll
  for (int mi = 0; mi < 2; ++mi) {
#pragma unroll
    for (int ni = 0; ni < 4; ++ni) {
      int n = n0 + ni * 16 + nl;
      float bias = b_ih[n];
#pragma unroll
      for (int r = 0; r < 8; ++r) {
        int m = m0 + mi * 16 + r + 8 * half;  // C/D: VGPR r -> M = r + 8*half
        x_ih[(size_t)m * G4 + n] = acc[mi][ni][r] + bias;
      }
    }
  }
}

// ---------------- Phase 3A: gates GEMM + LSTM cell (one timestep) ----------
// grid = 1024/32 WGs; WG owns 32 hidden units j; 8 waves -> 4 gates x 2 tiles.
// The bf16 mem matrix (16x1024 = 32 KB) is staged into LDS once per WG with
// gfx1250 async global->LDS copies, then all A-fragments come from LDS.

__global__ void k_step_gates(const bf16_t* __restrict__ mem_bf,   // MP x H (in)
                             const bf16_t* __restrict__ whh,      // 4H x H
                             const float* __restrict__ x_ih,      // BS x 4H
                             const float* __restrict__ b_hh,
                             const float* __restrict__ thr_p,
                             float* __restrict__ syn,              // B x H
                             float* __restrict__ mem,              // B x H
                             bf16_t* __restrict__ mem_bf_out,      // MP x H (out)
                             bf16_t* __restrict__ spk_bf,          // MP x H
                             int t) {
  __shared__ alignas(16) bf16_t a_smem[MP * H];  // 32 KB staged A matrix
  __shared__ float lds[8][16][16];               // 8 output tiles, 8 KB

  int lane = threadIdx.x & 31, w = threadIdx.x >> 5;

  // stage mem_bf -> LDS: 32 KB = 2048 x 16B chunks, 8 chunks per thread
#pragma unroll
  for (int it = 0; it < 8; ++it) {
    int elem = (threadIdx.x + 256 * it) * 8;  // 8 bf16 = 16 B per chunk
    unsigned lds_off = (unsigned)(size_t)&a_smem[elem];
    async_copy_b128(lds_off, mem_bf + elem);
  }
  wait_asynccnt0();
  __syncthreads();

  int gate = w >> 1, jh = w & 1;
  int jblk = blockIdx.x * 32;
  int n_glob = gate * H + jblk + jh * 16;

  v8f acc = {};
  for (int k = 0; k < H; k += 32) {
    if (k + 32 < H)
      __builtin_prefetch(whh + (size_t)(n_glob + (lane & 15)) * H + k + 32, 0, 1);
    v16bf a = load_a_frag(a_smem, H, 0, k, lane);   // LDS; rows 8..15 zero pad
    v16bf b = load_b_frag(whh, H, n_glob, k, lane);
    acc = wmma_bf16(a, b, acc);
  }
  int half = lane >> 4, n = lane & 15;
#pragma unroll
  for (int r = 0; r < 8; ++r) lds[w][r + 8 * half][n] = acc[r];
  __syncthreads();

  // elementwise cell update: 256 threads -> (m 0..7) x (jl 0..31)
  int m = threadIdx.x >> 5, jl = threadIdx.x & 31;
  int j = jblk + jl;
  int tile = jl >> 4, nn = jl & 15;
  float thr = thr_p[0];
  size_t row = (size_t)m * S + t;  // x row (b,s) -> m*S + t
  float gi = lds[0 + tile][m][nn] + x_ih[row * G4 + 0 * H + j] + b_hh[0 * H + j];
  float gf = lds[2 + tile][m][nn] + x_ih[row * G4 + 1 * H + j] + b_hh[1 * H + j];
  float gg = lds[4 + tile][m][nn] + x_ih[row * G4 + 2 * H + j] + b_hh[2 * H + j];
  float go = lds[6 + tile][m][nn] + x_ih[row * G4 + 3 * H + j] + b_hh[3 * H + j];

  int idx = m * H + j;
  float mem_old = mem[idx];
  float reset = (mem_old - thr > 0.0f) ? 1.0f : 0.0f;       // detached reset
  float c = sigmoidf_(gf) * syn[idx] + sigmoidf_(gi) * tanhf(gg);
  float hh = sigmoidf_(go) * tanhf(c);
  float mem_new = hh - reset * thr;
  float spk = (mem_new - thr > 0.0f) ? 1.0f : 0.0f;         // atan fwd = step

  syn[idx] = c;
  mem[idx] = mem_new;
  mem_bf_out[idx] = (bf16_t)mem_new;  // rows 8..15 of mem_bf_out stay zero
  spk_bf[idx] = (bf16_t)spk;          // exact in bf16
}

// ---------------- Phase 3B: out = Dyt(spk @ w_fc^T + b_fc + x_t) -----------
// grid = 1024/128 WGs, 8 waves, one 16x16 tile each.

__global__ void k_step_out(const bf16_t* __restrict__ spk_bf,   // MP x H
                           const bf16_t* __restrict__ wfc,      // H x H
                           const float* __restrict__ b_fc,
                           const float* __restrict__ x,         // B x S x D
                           const float* __restrict__ alpha_p,
                           const float* __restrict__ dyt_w,
                           const float* __restrict__ dyt_b,
                           float* __restrict__ out,             // B x S x H
                           int t) {
  int lane = threadIdx.x & 31, w = threadIdx.x >> 5;
  int n0 = blockIdx.x * 128 + w * 16;
  v8f acc = {};
  for (int k = 0; k < H; k += 32) {
    if (k + 32 < H)
      __builtin_prefetch(wfc + (size_t)(n0 + (lane & 15)) * H + k + 32, 0, 1);
    v16bf a = load_a_frag(spk_bf, H, 0, k, lane);
    v16bf b = load_b_frag(wfc, H, n0, k, lane);
    acc = wmma_bf16(a, b, acc);
  }
  int half = lane >> 4, n = n0 + (lane & 15);
  if (half == 0) {  // lanes 16..31 hold pad rows M=8..15
    float alpha = alpha_p[0], wq = dyt_w[n], bq = dyt_b[n], bias = b_fc[n];
#pragma unroll
    for (int r = 0; r < 8; ++r) {
      int m = r;  // M = r + 8*half, half==0
      float val = acc[r] + bias + x[((size_t)m * S + t) * D + n];
      out[((size_t)m * S + t) * H + n] = wq * tanhf(alpha * val) + bq;
    }
  }
}

// ---------------- Phase 4: append final (syn, mem) -------------------------

__global__ void k_final(const float* __restrict__ syn,
                        const float* __restrict__ mem,
                        float* __restrict__ out_tail) {
  int i = blockIdx.x * blockDim.x + threadIdx.x;
  if (i < Bb * H) {
    out_tail[i] = syn[i];
    out_tail[Bb * H + i] = mem[i];
  }
}

// ---------------------------------------------------------------------------

extern "C" void kernel_launch(void* const* d_in, const int* in_sizes, int n_in,
                              void* d_out, int out_size, void* d_ws,
                              size_t ws_size, hipStream_t stream) {
  const float* x     = (const float*)d_in[0];
  const float* w_ih  = (const float*)d_in[1];
  const float* w_hh  = (const float*)d_in[2];
  const float* b_ih  = (const float*)d_in[3];
  const float* b_hh  = (const float*)d_in[4];
  const float* w_fc  = (const float*)d_in[5];
  const float* b_fc  = (const float*)d_in[6];
  const float* thr   = (const float*)d_in[7];
  const float* alpha = (const float*)d_in[8];
  const float* dyt_w = (const float*)d_in[9];
  const float* dyt_b = (const float*)d_in[10];

  // workspace carve-up (256B aligned)
  char* ws = (char*)d_ws;
  size_t off = 0;
  auto carve = [&](size_t bytes) {
    void* p = ws + off;
    off = (off + bytes + 255) & ~size_t(255);
    return p;
  };
  bf16_t* xb    = (bf16_t*)carve((size_t)BS * D * 2);
  bf16_t* wihb  = (bf16_t*)carve((size_t)G4 * D * 2);
  bf16_t* whhb  = (bf16_t*)carve((size_t)G4 * H * 2);
  bf16_t* wfcb  = (bf16_t*)carve((size_t)H * H * 2);
  float*  x_ih  = (float*)carve((size_t)BS * G4 * 4);
  float*  syn   = (float*)carve((size_t)Bb * H * 4);
  float*  mem   = (float*)carve((size_t)Bb * H * 4);
  bf16_t* mbf0  = (bf16_t*)carve((size_t)MP * H * 2);  // ping
  bf16_t* mbf1  = (bf16_t*)carve((size_t)MP * H * 2);  // pong
  bf16_t* spkb  = (bf16_t*)carve((size_t)MP * H * 2);
  (void)ws_size; (void)in_sizes; (void)n_in; (void)out_size;

  // zero recurrent state (syn..spkb is one contiguous, 256B-aligned region)
  {
    size_t bytes = (size_t)(2 * Bb * H * 4) + (size_t)(3 * MP * H * 2);
    int n_words = (int)(bytes / 4);
    k_zero_u32<<<(n_words + 255) / 256, 256, 0, stream>>>((unsigned int*)syn,
                                                          n_words);
  }

  // bf16 conversions
  k_f32_to_bf16<<<(BS * D + 255) / 256, 256, 0, stream>>>(x, xb, BS * D);
  k_f32_to_bf16<<<(G4 * D + 255) / 256, 256, 0, stream>>>(w_ih, wihb, G4 * D);
  k_f32_to_bf16<<<(G4 * H + 255) / 256, 256, 0, stream>>>(w_hh, whhb, G4 * H);
  k_f32_to_bf16<<<(H * H + 255) / 256, 256, 0, stream>>>(w_fc, wfcb, H * H);

  // big parallel GEMM: x_ih
  dim3 g2(G4 / 256, BS / 64);
  k_gemm_xih<<<g2, 256, 0, stream>>>(xb, wihb, b_ih, x_ih);

  // sequential scan
  float* out = (float*)d_out;
  for (int t = 0; t < S; ++t) {
    bf16_t* m_in  = (t & 1) ? mbf1 : mbf0;
    bf16_t* m_out = (t & 1) ? mbf0 : mbf1;
    k_step_gates<<<H / 32, 256, 0, stream>>>(m_in, whhb, x_ih, b_hh, thr, syn,
                                             mem, m_out, spkb, t);
    k_step_out<<<H / 128, 256, 0, stream>>>(spkb, wfcb, b_fc, x, alpha, dyt_w,
                                            dyt_b, out, t);
  }

  // final syn / mem
  k_final<<<(Bb * H + 255) / 256, 256, 0, stream>>>(
      syn, mem, out + (size_t)BS * H);
}
